// DifferentiableRobustMVOLayer_27650999452176
// MI455X (gfx1250) — matrix operand order repaired and checked by reading, not compiled, not run on hardware
//
#include <hip/hip_runtime.h>

typedef __attribute__((ext_vector_type(16))) _Float16 v16h;
typedef __attribute__((ext_vector_type(8)))  _Float16 v8h;
typedef __attribute__((ext_vector_type(8)))  float    v8f;
typedef __attribute__((ext_vector_type(4)))  unsigned int v4u;

#define NDIM   256
#define BATCH  512
#define PITCH_H 264              // f16 elements per padded LDS row
#define PITCH_B (PITCH_H * 2)    // 528 bytes = 33*16 -> 16B aligned, bank-skewed
#define NITERS 400
#define LAMBD  1.0f
#define KAPPA  0.1f

__device__ __forceinline__ v16h cat8(v8h lo, v8h hi) {
    return __builtin_shufflevector(lo, hi, 0,1,2,3,4,5,6,7,8,9,10,11,12,13,14,15);
}

// ---------------------------------------------------------------------------
// Kernel 1: per (batch, which) compute G = M^T M in f16 via WMMA.
// Stage M transposed (Mt[j][i]) in padded LDS so both A- and B-fragments are
// contiguous 16B LDS reads matching the CDNA5 16-bit A 16x32 / B 32x16 layouts.
// Also emits ||M||_F^2 (needed for the PGD step size).
// ---------------------------------------------------------------------------
extern "C" __global__ void __launch_bounds__(512)
gram_f16_wmma(const float* __restrict__ U, const float* __restrict__ A,
              _Float16* __restrict__ G, float* __restrict__ fro)
{
    extern __shared__ char smem[];
    _Float16* Mt  = (_Float16*)smem;                    // [256][PITCH_H], transposed
    float*    red = (float*)(smem + NDIM * PITCH_B);    // 16 wave partials

    const int tid   = threadIdx.x;
    const int lane  = tid & 31;
    const int wave  = tid >> 5;
    const int b     = blockIdx.x >> 1;
    const int which = blockIdx.x & 1;

    const float* src = (which ? A : U) + (size_t)b * (NDIM * NDIM);
    _Float16* gdst = G + ((size_t)which * BATCH + b) * (NDIM * NDIM);

    // stage f32 -> f16 transposed, accumulate Frobenius^2
    float acc = 0.f;
#pragma unroll 4
    for (int t = 0; t < (NDIM * NDIM) / 512; ++t) {
        int idx = t * 512 + tid;
        int i = idx >> 8;
        int j = idx & (NDIM - 1);
        float v = src[idx];
        acc += v * v;
        Mt[j * PITCH_H + i] = (_Float16)v;
    }
#pragma unroll
    for (int mk = 16; mk >= 1; mk >>= 1) acc += __shfl_xor(acc, mk, 32);
    if (lane == 0) red[wave] = acc;
    __syncthreads();
    if (tid == 0) {
        float s = 0.f;
        for (int w2 = 0; w2 < 16; ++w2) s += red[w2];
        fro[which * BATCH + b] = s;
    }

    // G[j][k] = sum_i Mt[j][i]*Mt[k][i]; wave = row-block jt, loop col-blocks kt
    const int jt   = wave;
    const int m    = lane & 15;
    const int half = lane >> 4;
    const char* arow = (const char*)Mt + (jt * 16 + m) * PITCH_B;
    for (int kt = 0; kt < 16; ++kt) {
        const char* brow = (const char*)Mt + (kt * 16 + m) * PITCH_B;
        v8f c = {};
#pragma unroll
        for (int kc = 0; kc < 8; ++kc) {
            // A-frag slots: K = {half*8+0..7, 16+half*8+0..7} within the 32-chunk
            v8h a0 = *(const v8h*)(arow + kc * 64 + half * 16);
            v8h a1 = *(const v8h*)(arow + kc * 64 + 32 + half * 16);
            // B-frag slots: K = {half*8+0..7, 16+half*8+0..7} within the 32-chunk
            v8h b0 = *(const v8h*)(brow + kc * 64 + half * 16);
            v8h b1 = *(const v8h*)(brow + kc * 64 + 32 + half * 16);
            c = __builtin_amdgcn_wmma_f32_16x16x32_f16(false, cat8(a0, a1),
                                                       false, cat8(b0, b1),
                                                       (short)0, c, false, false);
        }
        // D layout: lane L, vgpr r -> (row = r + 8*(L/16), col = L%16)
        int col  = kt * 16 + m;
        int rowb = jt * 16 + half * 8;
#pragma unroll
        for (int r = 0; r < 8; ++r)
            gdst[(rowb + r) * NDIM + col] = (_Float16)c[r];
    }
}

// ---------------------------------------------------------------------------
// Kernel 2: one workgroup per batch. G1/G2 staged through LDS (f16, padded),
// then each wave's A-fragments (loop-invariant across all 400 iterations) are
// hoisted into 128 VGPRs. LDS limits us to 1 workgroup/WGP anyway, so the
// extra VGPRs cost nothing in occupancy. Per iteration only the broadcast
// w-fragment is re-read from LDS.
// ---------------------------------------------------------------------------
extern "C" __global__ void __launch_bounds__(512)
pgd_wmma(const float* __restrict__ mu, const _Float16* __restrict__ G,
         const float* __restrict__ fro, float* __restrict__ out)
{
    extern __shared__ char smem2[];
    _Float16* G1  = (_Float16*)smem2;
    _Float16* G2  = (_Float16*)(smem2 + NDIM * PITCH_B);
    float*    wv  = (float*)(smem2 + 2 * NDIM * PITCH_B);      // w (f32)
    _Float16* wh  = (_Float16*)((char*)wv + NDIM * 4);         // w (f16)
    float*    g1w = (float*)((char*)wh + NDIM * 2);
    float*    g2w = g1w + NDIM;
    float*    muL = g2w + NDIM;

    const int tid  = threadIdx.x;
    const int lane = tid & 31;
    const int wave = tid >> 5;
    const int b    = blockIdx.x;

    // stage G1/G2 (f16, row-major in ws) into padded LDS with b128 copies
    const v4u* s1 = (const v4u*)(G + (size_t)b * (NDIM * NDIM));
    const v4u* s2 = (const v4u*)(G + ((size_t)BATCH + b) * (NDIM * NDIM));
#pragma unroll 4
    for (int t = 0; t < 16; ++t) {
        int idx = t * 512 + tid;
        int row = idx >> 5;
        int ch  = idx & 31;
        *(v4u*)((char*)G1 + row * PITCH_B + ch * 16) = s1[idx];
        *(v4u*)((char*)G2 + row * PITCH_B + ch * 16) = s2[idx];
    }
    if (tid < NDIM) {
        muL[tid] = mu[(size_t)b * NDIM + tid];
        wv[tid]  = 1.0f / NDIM;
    }
    const float froU2 = fro[b];
    const float froA  = sqrtf(fro[BATCH + b]);
    const float step  = 1.0f / (LAMBD * froU2 + KAPPA * froA + 1.0f);
    __syncthreads();

    const int jt   = wave;                 // 16 waves <-> 16 row-blocks
    const int m    = lane & 15;
    const int half = lane >> 4;
    const char* arow1 = (const char*)G1 + (jt * 16 + m) * PITCH_B;
    const char* arow2 = (const char*)G2 + (jt * 16 + m) * PITCH_B;

    // hoist all A-fragments for this wave's row-block into registers:
    // 2 matrices x 8 K-chunks x v16h (8 VGPRs) = 128 VGPRs per lane.
    v16h a1f[8], a2f[8];
#pragma unroll
    for (int kc = 0; kc < 8; ++kc) {
        a1f[kc] = cat8(*(const v8h*)(arow1 + kc * 64 + half * 16),
                       *(const v8h*)(arow1 + kc * 64 + 32 + half * 16));
        a2f[kc] = cat8(*(const v8h*)(arow2 + kc * 64 + half * 16),
                       *(const v8h*)(arow2 + kc * 64 + 32 + half * 16));
    }

    for (int it = 0; it < NITERS; ++it) {
        if (tid < NDIM) wh[tid] = (_Float16)wv[tid];
        __syncthreads();

        // g1w/g2w row-block via 8 chained WMMAs each (w replicated across N);
        // only the w B-fragment comes from LDS (broadcast reads).
        v8f c1 = {}, c2 = {};
#pragma unroll
        for (int kc = 0; kc < 8; ++kc) {
            const char* wp = (const char*)wh + kc * 64 + half * 16;
            v16h bv = cat8(*(const v8h*)wp, *(const v8h*)(wp + 32));
            c1 = __builtin_amdgcn_wmma_f32_16x16x32_f16(false, a1f[kc], false, bv,
                                                        (short)0, c1, false, false);
            c2 = __builtin_amdgcn_wmma_f32_16x16x32_f16(false, a2f[kc], false, bv,
                                                        (short)0, c2, false, false);
        }
        if (m == 0) {  // lanes 0 and 16 hold column 0 of D
            int rowb = jt * 16 + half * 8;
#pragma unroll
            for (int r = 0; r < 8; ++r) {
                g1w[rowb + r] = c1[r];
                g2w[rowb + r] = c2[r];
            }
        }
        __syncthreads();

        // gradient + simplex projection on wave 0 (8 entries per lane)
        if (wave == 0) {
            float wd = 0.f;
#pragma unroll
            for (int t2 = 0; t2 < 8; ++t2) {
                int i = lane + 32 * t2;
                wd += wv[i] * g2w[i];
            }
#pragma unroll
            for (int mk = 16; mk >= 1; mk >>= 1) wd += __shfl_xor(wd, mk, 32);
            float nrm = sqrtf(fmaxf(wd, 1e-12f));
            float kn  = KAPPA / nrm;

            float vloc[8];
            float sum_v = 0.f, max_v = -3.0e38f;
#pragma unroll
            for (int t2 = 0; t2 < 8; ++t2) {
                int i = lane + 32 * t2;
                float g = -muL[i] + LAMBD * g1w[i] + kn * g2w[i];
                float x = wv[i] - step * g;
                vloc[t2] = x;
                sum_v += x;
                max_v = fmaxf(max_v, x);
            }
#pragma unroll
            for (int mk = 16; mk >= 1; mk >>= 1) {
                sum_v += __shfl_xor(sum_v, mk, 32);
                max_v = fmaxf(max_v, __shfl_xor(max_v, mk, 32));
            }
            // bisection for tau: f(tau)=sum(max(v-tau,0))-1, monotone decreasing
            float lo = (sum_v - 1.0f) * (1.0f / NDIM);  // f(lo) >= 0
            float hi = max_v;                            // f(hi) = -1
#pragma unroll 1
            for (int bs = 0; bs < 28; ++bs) {
                float tau = 0.5f * (lo + hi);
                float s = 0.f;
#pragma unroll
                for (int t2 = 0; t2 < 8; ++t2) s += fmaxf(vloc[t2] - tau, 0.f);
#pragma unroll
                for (int mk = 16; mk >= 1; mk >>= 1) s += __shfl_xor(s, mk, 32);
                if (s > 1.0f) lo = tau; else hi = tau;
            }
            float tau = 0.5f * (lo + hi);
#pragma unroll
            for (int t2 = 0; t2 < 8; ++t2) {
                int i = lane + 32 * t2;
                wv[i] = fmaxf(vloc[t2] - tau, 0.f);
            }
        }
        __syncthreads();
    }

    // final clamp (already >=0) + renormalize, write output
    if (wave == 0) {
        float s = 0.f;
#pragma unroll
        for (int t2 = 0; t2 < 8; ++t2) s += wv[lane + 32 * t2];
#pragma unroll
        for (int mk = 16; mk >= 1; mk >>= 1) s += __shfl_xor(s, mk, 32);
        float inv = 1.0f / (s + 1e-12f);
#pragma unroll
        for (int t2 = 0; t2 < 8; ++t2) {
            int i = lane + 32 * t2;
            out[(size_t)b * NDIM + i] = wv[i] * inv;
        }
    }
}

extern "C" void kernel_launch(void* const* d_in, const int* in_sizes, int n_in,
                              void* d_out, int out_size, void* d_ws, size_t ws_size,
                              hipStream_t stream) {
    (void)in_sizes; (void)n_in; (void)out_size; (void)ws_size;
    const float* mu = (const float*)d_in[0];
    const float* U  = (const float*)d_in[1];
    const float* A  = (const float*)d_in[2];
    float* out = (float*)d_out;

    // ws layout: [G f16: 2*512*256*256] [fro f32: 2*512]
    _Float16* G  = (_Float16*)d_ws;
    float*   fro = (float*)((char*)d_ws + (size_t)2 * BATCH * NDIM * NDIM * 2);

    size_t lds1 = (size_t)NDIM * PITCH_B + 64;                           // ~132 KB
    gram_f16_wmma<<<dim3(BATCH * 2), dim3(512), lds1, stream>>>(U, A, G, fro);

    size_t lds2 = (size_t)2 * NDIM * PITCH_B + NDIM * 4 + NDIM * 2 + 3 * NDIM * 4; // ~268.5 KB
    pgd_wmma<<<dim3(BATCH), dim3(512), lds2, stream>>>(mu, G, fro, out);
}
